// MultiHeadAttention_8426725835273
// MI455X (gfx1250) — matrix operand order repaired and checked by reading, not compiled, not run on hardware
//
#include <hip/hip_runtime.h>
#include <hip/hip_bf16.h>

// MultiHeadAttention (block-local), MI455X / gfx1250, wave32 bf16 WMMA path
// with GLOBAL_LOAD_ASYNC_TO_LDS staging (ASYNCcnt) and double-buffered LDS.
// B=4 T=4096 C=1024 H=16 NBLOCKS=32 BSIZE=128 HD=64

#define B_  4
#define T_  4096
#define C_  1024
#define H_  16
#define NB_ 32
#define BS_ 128
#define HD_ 64

typedef __attribute__((ext_vector_type(16))) __bf16 v16bf;
typedef __attribute__((ext_vector_type(8)))  float  v8f;

__device__ __forceinline__ unsigned short f2bf(float f) {
  unsigned int u = __float_as_uint(f);
  u += 0x7FFFu + ((u >> 16) & 1u);          // round-to-nearest-even
  return (unsigned short)(u >> 16);
}
__device__ __forceinline__ unsigned int pk2(float a, float b) {
  return (unsigned int)f2bf(a) | ((unsigned int)f2bf(b) << 16);
}
__device__ __forceinline__ v8f v8f_zero() {
  v8f z;
  #pragma unroll
  for (int i = 0; i < 8; ++i) z[i] = 0.0f;
  return z;
}

// Low 32 bits of a generic pointer to LDS = wave-relative LDS byte address
// (ISA 10.2: the LDS aperture uses the low 32 address bits).
__device__ __forceinline__ unsigned lds_off(const void* p) {
  return (unsigned)(unsigned long long)p;
}

// CDNA5 async copy: 32B / 64B global -> LDS, tracked by ASYNCcnt.
// INST_OFFSET applies to BOTH the LDS and global addresses (ISA 10. async),
// so contiguous copies need only one LDS-addr VGPR and one 64-bit vaddr.
__device__ __forceinline__ void async_cp32(unsigned ldsoff, const void* g) {
  unsigned long long ga = (unsigned long long)g;
  asm volatile(
      "global_load_async_to_lds_b128 %0, %1, off\n\t"
      "global_load_async_to_lds_b128 %0, %1, off offset:16"
      :: "v"(ldsoff), "v"(ga) : "memory");
}
__device__ __forceinline__ void async_cp64(unsigned ldsoff, const void* g) {
  unsigned long long ga = (unsigned long long)g;
  asm volatile(
      "global_load_async_to_lds_b128 %0, %1, off\n\t"
      "global_load_async_to_lds_b128 %0, %1, off offset:16\n\t"
      "global_load_async_to_lds_b128 %0, %1, off offset:32\n\t"
      "global_load_async_to_lds_b128 %0, %1, off offset:48"
      :: "v"(ldsoff), "v"(ga) : "memory");
}
#define WAIT_ASYNC_0() asm volatile("s_wait_asynccnt 0x0" ::: "memory")
#define WAIT_ASYNC_4() asm volatile("s_wait_asynccnt 0x4" ::: "memory")

union FragU { uint4 q[2]; v16bf v; };

// A-matrix fragment (16x32 bf16): lane&15 = M row, lane>>4 = K half;
// per lane 16 consecutive K values.
__device__ __forceinline__ v16bf ldsA(const unsigned short* s, int off) {
  FragU f;
  f.q[0] = *(const uint4*)(s + off);
  f.q[1] = *(const uint4*)(s + off + 8);
  return f.v;
}
// B-matrix fragment (32x16 bf16): lane&15 = N column, lane>>4 = K half;
// per lane 16 consecutive K values.
__device__ __forceinline__ v16bf ldsB(const unsigned short* s, int off) {
  FragU f;
  f.q[0] = *(const uint4*)(s + off);
  f.q[1] = *(const uint4*)(s + off + 8);
  return f.v;
}

// ---------------------------------------------------------------------------
// One-shot f32 -> bf16 convert (x and the four weight matrices). These are
// tiny streaming passes; doing the conversion once removes the redundant
// per-tile VALU packing from the GEMM inner loops.
// ---------------------------------------------------------------------------
__global__ __launch_bounds__(256) void cvt_bf16(const float* __restrict__ src,
                                                unsigned short* __restrict__ dst,
                                                int n) {
  const int i = (blockIdx.x * 256 + threadIdx.x) * 8;
  if (i + 8 <= n) {
    float4 a = ((const float4*)(src + i))[0];
    float4 b = ((const float4*)(src + i))[1];
    uint4 p;
    p.x = pk2(a.x, a.y); p.y = pk2(a.z, a.w);
    p.z = pk2(b.x, b.y); p.w = pk2(b.z, b.w);
    *(uint4*)(dst + i) = p;
  }
}

// ---------------------------------------------------------------------------
// GEMM: Out[m,n] = (sum_k A[m,k]*W[n,k] + bias[n]) * scale
// A, W bf16 row-major (K inner). Tile 128x128, 8 waves (4x2), K-step 32.
// LDS staging via async-to-LDS, double buffered: issue tile k+1, retire
// tile k with s_wait_asynccnt 4 (async loads complete in order).
// ---------------------------------------------------------------------------
template <bool OUT_F32>
__global__ __launch_bounds__(256) void gemm_bf16(
    const unsigned short* __restrict__ A, const unsigned short* __restrict__ W,
    const float* __restrict__ bias, void* __restrict__ Out, float scale,
    int M, int N, int K) {
  __shared__ unsigned short sA[2 * 128 * 40];   // 2 bufs, rows pad to 40
  __shared__ unsigned short sB[2 * 128 * 40];

  const int tid  = threadIdx.x;
  const int m0   = blockIdx.x * 128;
  const int n0   = blockIdx.y * 128;
  const int row  = tid >> 1, half = tid & 1;          // staging: 2 thr/row
  const int lane = tid & 31, wv = tid >> 5;
  const int wm   = wv & 3,  wn = wv >> 2;             // 4x2 wave grid
  const int l15  = lane & 15, lh = lane >> 4;

  const unsigned short* srcA = A + (size_t)(m0 + row) * K + half * 16;
  const unsigned short* srcB = W + (size_t)(n0 + row) * K + half * 16;
  const unsigned dA0 = lds_off(sA + row * 40 + half * 16);
  const unsigned dB0 = lds_off(sB + row * 40 + half * 16);

  v8f acc[2][4];
  #pragma unroll
  for (int mi = 0; mi < 2; ++mi)
    #pragma unroll
    for (int ni = 0; ni < 4; ++ni) acc[mi][ni] = v8f_zero();

  const int KT = K >> 5;                 // 32-wide K tiles
  // prologue: tile 0 -> buffer 0 (4 async ops per thread)
  async_cp32(dA0, srcA);
  async_cp32(dB0, srcB);

  for (int kt = 0; kt < KT; ++kt) {
    const int cur = kt & 1;
    if (kt + 1 < KT) {                   // issue tile kt+1 into other buffer
      const unsigned boff = (cur ^ 1) * (128 * 40 * 2);  // bytes
      async_cp32(dA0 + boff, srcA + (kt + 1) * 32);
      async_cp32(dB0 + boff, srcB + (kt + 1) * 32);
      WAIT_ASYNC_4();                    // retire tile kt (in-order)
    } else {
      WAIT_ASYNC_0();
    }
    __syncthreads();                     // tile kt visible to all waves

    const unsigned short* cA = sA + cur * (128 * 40);
    const unsigned short* cB = sB + cur * (128 * 40);
    v16bf af[2], bfr[4];
    #pragma unroll
    for (int mi = 0; mi < 2; ++mi)
      af[mi] = ldsA(cA, (32 * wm + 16 * mi + l15) * 40 + lh * 16);
    #pragma unroll
    for (int ni = 0; ni < 4; ++ni)
      bfr[ni] = ldsB(cB, (64 * wn + 16 * ni + l15) * 40 + lh * 16);

    #pragma unroll
    for (int mi = 0; mi < 2; ++mi)
      #pragma unroll
      for (int ni = 0; ni < 4; ++ni)
        acc[mi][ni] = __builtin_amdgcn_wmma_f32_16x16x32_bf16(
            false, af[mi], false, bfr[ni], (short)0, acc[mi][ni], false, false);

    __syncthreads();                     // all reads of buf cur done
  }

  // epilogue: D layout: lane&15 = N col, VGPR j + 8*(lane>>4) = M row
  #pragma unroll
  for (int mi = 0; mi < 2; ++mi)
    #pragma unroll
    for (int ni = 0; ni < 4; ++ni) {
      const int n = n0 + 64 * wn + 16 * ni + l15;
      const float bval = bias ? bias[n] : 0.0f;
      #pragma unroll
      for (int j = 0; j < 8; ++j) {
        const int m = m0 + 32 * wm + 16 * mi + (lh << 3) + j;
        const float v = (acc[mi][ni][j] + bval) * scale;
        if (OUT_F32)
          ((float*)Out)[(size_t)m * N + n] = v;
        else
          ((unsigned short*)Out)[(size_t)m * N + n] = f2bf(v);
      }
    }
}

// ---------------------------------------------------------------------------
// Block-local attention. One workgroup per (head, block, batch).
// Q,K,V bf16 in (B,T,C) layout, c = h*64+d; HD^-0.25 already folded into q,k.
// Each of 8 waves owns a 16-row strip of the 128x128 score matrix.
// ---------------------------------------------------------------------------
__global__ __launch_bounds__(256) void blockattn(
    const unsigned short* __restrict__ Q, const unsigned short* __restrict__ Kb,
    const unsigned short* __restrict__ V, unsigned short* __restrict__ O) {
  // Q rows stride 72 at [0, 9216); K rows stride 72 at [9216, 18432);
  // P (128x128, stride 136) aliased over the whole region after S is done.
  __shared__ unsigned short sQK[18432];
  __shared__ unsigned short sV[64 * 136];   // V^T: [n][k], stride 136

  const int h  = blockIdx.x, nb = blockIdx.y, b = blockIdx.z;
  const int tid  = threadIdx.x;
  const int lane = tid & 31, wv = tid >> 5;
  const int l15  = lane & 15, lh = lane >> 4;
  const size_t base = ((size_t)(b * T_ + nb * BS_) * C_) + (size_t)h * HD_;

  // ---- stage Q and K row-major via async-to-LDS (64B per thread each) ----
  {
    const int row = tid >> 1, half = tid & 1;
    const size_t g = base + (size_t)row * C_ + half * 32;
    async_cp64(lds_off(sQK + row * 72 + half * 32), Q + g);
    async_cp64(lds_off(sQK + 9216 + row * 72 + half * 32), Kb + g);
  }
  // ---- stage V transposed: sV[n][k] = V[k][n] ----
  #pragma unroll
  for (int i = 0; i < 8; ++i) {
    const int cid = tid + i * 256;           // 2048 uint2 chunks
    const int k = cid >> 4, n4 = (cid & 15) << 2;
    uint2 d = *(const uint2*)(V + base + (size_t)k * C_ + n4);
    sV[(n4 + 0) * 136 + k] = (unsigned short)(d.x & 0xffffu);
    sV[(n4 + 1) * 136 + k] = (unsigned short)(d.x >> 16);
    sV[(n4 + 2) * 136 + k] = (unsigned short)(d.y & 0xffffu);
    sV[(n4 + 3) * 136 + k] = (unsigned short)(d.y >> 16);
  }
  WAIT_ASYNC_0();
  __syncthreads();

  // ---- S = Q K^T for this wave's 16-row strip (8 N-tiles, K=64, 2 steps) ----
  v8f sacc[8];
  #pragma unroll
  for (int t = 0; t < 8; ++t) sacc[t] = v8f_zero();
  #pragma unroll
  for (int kk = 0; kk < 64; kk += 32) {
    v16bf aq = ldsA(sQK, (16 * wv + l15) * 72 + kk + lh * 16);
    #pragma unroll
    for (int t = 0; t < 8; ++t) {
      v16bf bk = ldsB(sQK + 9216, (16 * t + l15) * 72 + kk + lh * 16);
      sacc[t] = __builtin_amdgcn_wmma_f32_16x16x32_bf16(
          false, aq, false, bk, (short)0, sacc[t], false, false);
    }
  }
  __syncthreads();   // everyone done reading Q/K LDS -> safe to alias with P

  // ---- softmax per row; row = 16*wv + j + 8*(lane>>4); cols = 16*t + l15 ----
  #pragma unroll
  for (int j = 0; j < 8; ++j) {
    float mx = sacc[0][j];
    #pragma unroll
    for (int t = 1; t < 8; ++t) mx = fmaxf(mx, sacc[t][j]);
    #pragma unroll
    for (int m = 1; m < 16; m <<= 1) mx = fmaxf(mx, __shfl_xor(mx, m, 32));
    float e[8], sum = 0.0f;
    #pragma unroll
    for (int t = 0; t < 8; ++t) { e[t] = __expf(sacc[t][j] - mx); sum += e[t]; }
    #pragma unroll
    for (int m = 1; m < 16; m <<= 1) sum += __shfl_xor(sum, m, 32);
    const float inv = 1.0f / sum;
    const int rp = 16 * wv + (lh << 3) + j;  // this wave's private P rows
    #pragma unroll
    for (int t = 0; t < 8; ++t)
      sQK[rp * 136 + 16 * t + l15] = f2bf(e[t] * inv);
  }

  // ---- O = P V : 16x64 strip, K=128 in 4 steps, 4 N-tiles ----
  v8f oacc[4];
  #pragma unroll
  for (int nt = 0; nt < 4; ++nt) oacc[nt] = v8f_zero();
  #pragma unroll
  for (int kk = 0; kk < 128; kk += 32) {
    v16bf pa = ldsA(sQK, (16 * wv + l15) * 136 + kk + lh * 16);
    #pragma unroll
    for (int nt = 0; nt < 4; ++nt) {
      v16bf vf = ldsB(sV, (16 * nt + l15) * 136 + kk + lh * 16);
      oacc[nt] = __builtin_amdgcn_wmma_f32_16x16x32_bf16(
          false, pa, false, vf, (short)0, oacc[nt], false, false);
    }
  }
  // ---- store O as bf16 in (B,T,C) layout, c = h*64 + d ----
  #pragma unroll
  for (int nt = 0; nt < 4; ++nt)
    #pragma unroll
    for (int j = 0; j < 8; ++j) {
      const int r = 16 * wv + (lh << 3) + j;
      O[base + (size_t)r * C_ + nt * 16 + l15] = f2bf(oacc[nt][j]);
    }
}

// ---------------------------------------------------------------------------
extern "C" void kernel_launch(void* const* d_in, const int* in_sizes, int n_in,
                              void* d_out, int out_size, void* d_ws,
                              size_t ws_size, hipStream_t stream) {
  const float* x  = (const float*)d_in[0];
  const float* Wq = (const float*)d_in[1];
  const float* bq = (const float*)d_in[2];
  const float* Wk = (const float*)d_in[3];
  const float* Wv = (const float*)d_in[4];
  const float* bv = (const float*)d_in[5];
  const float* Wo = (const float*)d_in[6];
  const float* bo = (const float*)d_in[7];
  float* out = (float*)d_out;

  const size_t MT = (size_t)B_ * T_;   // 16384 rows
  const size_t CC = (size_t)C_ * C_;   // 1M weight elements
  unsigned short* xb  = (unsigned short*)d_ws;   // bf16 x      (32 MB)
  unsigned short* wqb = xb  + MT * C_;           // bf16 weights (2 MB each)
  unsigned short* wkb = wqb + CC;
  unsigned short* wvb = wkb + CC;
  unsigned short* wob = wvb + CC;
  unsigned short* qb  = wob + CC;                // bf16 q/k/v/o (32 MB each)
  unsigned short* kb  = qb  + MT * C_;
  unsigned short* vb  = kb  + MT * C_;
  unsigned short* ob  = vb  + MT * C_;

  // one-shot conversions (x: 16.7M elems, each W: 1M elems)
  cvt_bf16<<<(unsigned)(MT * C_ / 2048), 256, 0, stream>>>(x, xb, (int)(MT * C_));
  cvt_bf16<<<(unsigned)(CC / 2048), 256, 0, stream>>>(Wq, wqb, (int)CC);
  cvt_bf16<<<(unsigned)(CC / 2048), 256, 0, stream>>>(Wk, wkb, (int)CC);
  cvt_bf16<<<(unsigned)(CC / 2048), 256, 0, stream>>>(Wv, wvb, (int)CC);
  cvt_bf16<<<(unsigned)(CC / 2048), 256, 0, stream>>>(Wo, wob, (int)CC);

  const float scale = 0.35355339059327373f;      // 64^(-0.25)
  dim3 gg((unsigned)(MT / 128), C_ / 128);

  gemm_bf16<false><<<gg, 256, 0, stream>>>(xb, wqb, bq, qb, scale,
                                           (int)MT, C_, C_);
  gemm_bf16<false><<<gg, 256, 0, stream>>>(xb, wkb, nullptr, kb, scale,
                                           (int)MT, C_, C_);
  gemm_bf16<false><<<gg, 256, 0, stream>>>(xb, wvb, bv, vb, 1.0f,
                                           (int)MT, C_, C_);
  blockattn<<<dim3(H_, NB_, B_), 256, 0, stream>>>(qb, kb, vb, ob);
  gemm_bf16<true><<<gg, 256, 0, stream>>>(ob, wob, bo, out, 1.0f,
                                          (int)MT, C_, C_);
}
